// NT_Xent_49400713838965
// MI455X (gfx1250) — matrix operand order repaired and checked by reading, not compiled, not run on hardware
//
#include <hip/hip_runtime.h>
#include <hip/hip_bf16.h>

#define N_TOT 8192
#define BATCH 4096
#define DIM   128
#define NCHUNK 16
#define COLS_PER_CHUNK (N_TOT / NCHUNK)       // 512
#define TILES_PER_CHUNK (COLS_PER_CHUNK / 16) // 32

typedef __attribute__((ext_vector_type(16))) _Float16 v16h;
typedef __attribute__((ext_vector_type(8)))  _Float16 v8h;
typedef __attribute__((ext_vector_type(8)))  float    v8f;

union V16U { v16h v; v8h h[2]; };

constexpr float INV_T    = 2.0f;                      // 1 / 0.5 temperature
constexpr float LOG2E    = 1.44269504088896340736f;
constexpr float LN2      = 0.69314718055994530942f;
constexpr float EXPSCALE = INV_T * LOG2E;             // exp(x/T) = exp2(x*EXPSCALE)

// ---------------------------------------------------------------------------
// Kernel 1: L2-normalize each row of [z_i; z_j], emit f16 rows + self-dot
// (self-dot of the *quantized* row, used for the diagonal correction).
// ---------------------------------------------------------------------------
__global__ __launch_bounds__(DIM) void k_normalize(const float* __restrict__ zi,
                                                   const float* __restrict__ zj,
                                                   _Float16* __restrict__ zn,
                                                   float* __restrict__ selfdot) {
  const int row = blockIdx.x;
  const int t   = threadIdx.x;
  const float* src = (row < BATCH) ? (zi + (size_t)row * DIM)
                                   : (zj + (size_t)(row - BATCH) * DIM);
  float x  = src[t];
  float ss = x * x;
  #pragma unroll
  for (int m = 16; m >= 1; m >>= 1) ss += __shfl_xor(ss, m, 32);
  __shared__ float red[4];
  __shared__ float red2[4];
  if ((t & 31) == 0) red[t >> 5] = ss;
  __syncthreads();
  float tot = red[0] + red[1] + red[2] + red[3];
  float inv = 1.0f / fmaxf(sqrtf(tot), 1e-8f);
  _Float16 h = (_Float16)(x * inv);
  zn[(size_t)row * DIM + t] = h;
  float hf = (float)h;
  float sd = hf * hf;
  #pragma unroll
  for (int m = 16; m >= 1; m >>= 1) sd += __shfl_xor(sd, m, 32);
  if ((t & 31) == 0) red2[t >> 5] = sd;
  __syncthreads();
  if (t == 0) selfdot[row] = red2[0] + red2[1] + red2[2] + red2[3];
}

// ---------------------------------------------------------------------------
// Kernel 2: positive-pair logits pos[i] = dot(zn[i], zn[(i+B)%N]) / T
// ---------------------------------------------------------------------------
__global__ __launch_bounds__(DIM) void k_pos(const _Float16* __restrict__ zn,
                                             float* __restrict__ pos) {
  const int row = blockIdx.x;
  const int t   = threadIdx.x;
  const int p   = (row + BATCH) & (N_TOT - 1);
  float a = (float)zn[(size_t)row * DIM + t];
  float b = (float)zn[(size_t)p   * DIM + t];
  float d = a * b;
  #pragma unroll
  for (int m = 16; m >= 1; m >>= 1) d += __shfl_xor(d, m, 32);
  __shared__ float red[4];
  if ((t & 31) == 0) red[t >> 5] = d;
  __syncthreads();
  if (t == 0) pos[row] = (red[0] + red[1] + red[2] + red[3]) * INV_T;
}

// ---------------------------------------------------------------------------
// Kernel 3: fused Gram-matrix + sum-of-exp.  Each wave owns TWO 16-row strips
// (A operands resident: 64 VGPRs) and sweeps the column tiles of its chunk:
//   8 x v_wmma_f32_16x16x32_f16 per tile (two independent accumulator chains
//   sharing the B operand -> 2x arithmetic intensity on B loads)
//   16 x v_exp_f32 online accumulation (no max needed: |sim/T| <= 2)
// C/D layout: VGPR r <-> row r + 8*(lane>=16); lane%16 <-> column.
// ---------------------------------------------------------------------------
__global__ __launch_bounds__(256) void k_sim_lse(const _Float16* __restrict__ zn,
                                                 float* __restrict__ rowsum) {
  const int wave = threadIdx.x >> 5;
  const int lane = threadIdx.x & 31;
  const int half = lane >> 4;   // 0: lanes 0-15, 1: lanes 16-31
  const int l16  = lane & 15;

  const int pair = blockIdx.x * 8 + wave;           // 32-row pair index
  const int row0 = pair * 32;
  const int col0 = blockIdx.y * COLS_PER_CHUNK;

  // --- A operands: two 16-row strips; contiguous K=16 per lane half
  //     (lane<16: K+0..15, lane>=16: K+16..31)
  const _Float16* arow0 = zn + (size_t)(row0 + l16) * DIM + half * 16;
  const _Float16* arow1 = arow0 + (size_t)16 * DIM;
  v16h a0[4], a1[4];
  #pragma unroll
  for (int kb = 0; kb < 4; ++kb) {
    V16U u0, u1;
    u0.h[0] = *(const v8h*)(arow0 + kb * 32);
    u0.h[1] = *(const v8h*)(arow0 + kb * 32 + 8);
    u1.h[0] = *(const v8h*)(arow1 + kb * 32);
    u1.h[1] = *(const v8h*)(arow1 + kb * 32 + 8);
    a0[kb] = u0.v;
    a1[kb] = u1.v;
  }

  // --- B operand: per-lane column col0+l16; contiguous K=16 per lane half
  const _Float16* bptr = zn + (size_t)(col0 + l16) * DIM + half * 16;

  float s0[8], s1[8];
  #pragma unroll
  for (int r = 0; r < 8; ++r) { s0[r] = 0.0f; s1[r] = 0.0f; }

  for (int t = 0; t < TILES_PER_CHUNK; ++t) {
    v16h b0 = *(const v16h*)(bptr);
    v16h b1 = *(const v16h*)(bptr + 32);
    v16h b2 = *(const v16h*)(bptr + 64);
    v16h b3 = *(const v16h*)(bptr + 96);
    __builtin_prefetch(bptr + 16 * DIM, 0, 3);   // next tile -> WGP-scope prefetch

    v8f c0 = {};
    v8f c1 = {};
    c0 = __builtin_amdgcn_wmma_f32_16x16x32_f16(false, a0[0], false, b0, (short)0, c0, false, false);
    c1 = __builtin_amdgcn_wmma_f32_16x16x32_f16(false, a1[0], false, b0, (short)0, c1, false, false);
    c0 = __builtin_amdgcn_wmma_f32_16x16x32_f16(false, a0[1], false, b1, (short)0, c0, false, false);
    c1 = __builtin_amdgcn_wmma_f32_16x16x32_f16(false, a1[1], false, b1, (short)0, c1, false, false);
    c0 = __builtin_amdgcn_wmma_f32_16x16x32_f16(false, a0[2], false, b2, (short)0, c0, false, false);
    c1 = __builtin_amdgcn_wmma_f32_16x16x32_f16(false, a1[2], false, b2, (short)0, c1, false, false);
    c0 = __builtin_amdgcn_wmma_f32_16x16x32_f16(false, a0[3], false, b3, (short)0, c0, false, false);
    c1 = __builtin_amdgcn_wmma_f32_16x16x32_f16(false, a1[3], false, b3, (short)0, c1, false, false);

    #pragma unroll
    for (int r = 0; r < 8; ++r) {
      s0[r] += __builtin_amdgcn_exp2f(c0[r] * EXPSCALE);
      s1[r] += __builtin_amdgcn_exp2f(c1[r] * EXPSCALE);
    }

    bptr += 16 * DIM;
  }

  // Reduce each row's partial sum across its 16 lanes.
  #pragma unroll
  for (int r = 0; r < 8; ++r) {
    float v0 = s0[r];
    v0 += __shfl_xor(v0, 1, 16);
    v0 += __shfl_xor(v0, 2, 16);
    v0 += __shfl_xor(v0, 4, 16);
    v0 += __shfl_xor(v0, 8, 16);
    s0[r] = v0;
    float v1 = s1[r];
    v1 += __shfl_xor(v1, 1, 16);
    v1 += __shfl_xor(v1, 2, 16);
    v1 += __shfl_xor(v1, 4, 16);
    v1 += __shfl_xor(v1, 8, 16);
    s1[r] = v1;
  }
  if (l16 == 0) {
    float* dst0 = rowsum + (size_t)blockIdx.y * N_TOT + row0 + half * 8;
    float* dst1 = dst0 + 16;
    #pragma unroll
    for (int r = 0; r < 8; ++r) { dst0[r] = s0[r]; dst1[r] = s1[r]; }
  }
}

// ---------------------------------------------------------------------------
// Kernel 4: loss = mean( log(S_i - exp(self_i/T)) - pos_i )
// Fixed-order reductions -> deterministic.
// ---------------------------------------------------------------------------
__global__ __launch_bounds__(256) void k_finalize(const float* __restrict__ rowsum,
                                                  const float* __restrict__ selfdot,
                                                  const float* __restrict__ pos,
                                                  float* __restrict__ out) {
  float acc = 0.0f;
  for (int i = threadIdx.x; i < N_TOT; i += 256) {
    float S = 0.0f;
    #pragma unroll
    for (int ch = 0; ch < NCHUNK; ++ch) S += rowsum[(size_t)ch * N_TOT + i];
    S -= __builtin_amdgcn_exp2f(selfdot[i] * EXPSCALE);   // remove diagonal
    float lse = __builtin_amdgcn_logf(S) * LN2;           // natural log
    acc += lse - pos[i];
  }
  #pragma unroll
  for (int m = 16; m >= 1; m >>= 1) acc += __shfl_xor(acc, m, 32);
  __shared__ float red[8];
  if ((threadIdx.x & 31) == 0) red[threadIdx.x >> 5] = acc;
  __syncthreads();
  if (threadIdx.x == 0) {
    float tot = 0.0f;
    #pragma unroll
    for (int w = 0; w < 8; ++w) tot += red[w];
    out[0] = tot / (float)N_TOT;
  }
}

// ---------------------------------------------------------------------------
extern "C" void kernel_launch(void* const* d_in, const int* in_sizes, int n_in,
                              void* d_out, int out_size, void* d_ws, size_t ws_size,
                              hipStream_t stream) {
  const float* zi = (const float*)d_in[0];
  const float* zj = (const float*)d_in[1];

  char* ws = (char*)d_ws;
  _Float16* zn   = (_Float16*)ws;                                   // 2 MB
  char* p = ws + (size_t)N_TOT * DIM * sizeof(_Float16);
  float* rowsum  = (float*)p;  p += (size_t)NCHUNK * N_TOT * sizeof(float);  // 512 KB
  float* selfdot = (float*)p;  p += (size_t)N_TOT * sizeof(float);           // 32 KB
  float* pos     = (float*)p;                                                // 32 KB

  k_normalize<<<N_TOT, DIM, 0, stream>>>(zi, zj, zn, selfdot);
  k_pos<<<N_TOT, DIM, 0, stream>>>(zn, pos);
  dim3 grid(N_TOT / (32 * 8), NCHUNK);            // 32 x 16 blocks, 8 waves each
  k_sim_lse<<<grid, 256, 0, stream>>>(zn, rowsum);
  k_finalize<<<1, 256, 0, stream>>>(rowsum, selfdot, pos, (float*)d_out);
}
